// CrissCrossAttention_63196148793370
// MI455X (gfx1250) — compile-verified, matrix operand order checked by reading, not run on hardware
//
#include <hip/hip_runtime.h>
#include <hip/hip_bf16.h>

#define B_ 16
#define C_ 512
#define H_ 64
#define W_ 64
#define CQ_ 64
#define S_ (H_ * W_)
#define NEG_INF_ (-1e30f)

typedef __attribute__((ext_vector_type(16))) __bf16 v16bf;
typedef __attribute__((ext_vector_type(8)))  __bf16 v8bf;
typedef __attribute__((ext_vector_type(8)))  float  v8f;
typedef __attribute__((ext_vector_type(4)))  float  v4f;

__device__ __forceinline__ v16bf ld16_bf(const __bf16* p) {
  return *reinterpret_cast<const v16bf*>(p);
}

__device__ __forceinline__ v8f wmma_bf16(v16bf a, v16bf b, v8f c) {
  // emits v_wmma_f32_16x16x32_bf16
  return __builtin_amdgcn_wmma_f32_16x16x32_bf16(
      /*neg_a=*/false, a, /*neg_b=*/false, b,
      /*c_mod=*/(short)0, c, /*reuse_a=*/false, /*reuse_b=*/false);
}

__device__ __forceinline__ void zero_acc(v8f (&acc)[4]) {
#pragma unroll
  for (int j = 0; j < 4; ++j)
#pragma unroll
    for (int r = 0; r < 8; ++r) acc[j][r] = 0.0f;
}

// ---------------------------------------------------------------------------
// Kernel 0: pre-convert concatenated weights [Wq;Wk;Wv] (640 x 512) to bf16.
// ---------------------------------------------------------------------------
__global__ __launch_bounds__(256) void prep_weights_kernel(
    const float* __restrict__ Wq, const float* __restrict__ Wk,
    const float* __restrict__ Wv, __bf16* __restrict__ wt) {
  const int i = (blockIdx.x * 256 + threadIdx.x) * 4;
  const int row = i >> 9;
  const int col = i & 511;
  const float* src;
  if (row < 64)       src = Wq + (size_t)row * C_ + col;
  else if (row < 128) src = Wk + (size_t)(row - 64) * C_ + col;
  else                src = Wv + (size_t)(row - 128) * C_ + col;
  const v4f v = *reinterpret_cast<const v4f*>(src);
#pragma unroll
  for (int j = 0; j < 4; ++j) wt[i + j] = (__bf16)v[j];
}

// ---------------------------------------------------------------------------
// Kernel 1: fused QKV projection, double-buffered LDS.
// grid = (S/64 stiles, 10 otiles, B); block = 128 (4 waves).
// x chunk (32k x 64s) staged transposed (xs_t[buf][s][k], stride 40 ->
// conflict-free 16B ds reads). One barrier per K-step; next chunk's global
// loads issue before this chunk's WMMAs to hide latency behind matrix work.
// ---------------------------------------------------------------------------
__global__ __launch_bounds__(128) void qkv_proj_kernel(
    const float* __restrict__ x, const __bf16* __restrict__ wt,
    const float* __restrict__ bq, const float* __restrict__ bk,
    const float* __restrict__ bv,
    __bf16* __restrict__ qws, __bf16* __restrict__ kws,
    __bf16* __restrict__ vws, __bf16* __restrict__ vtws) {
  __shared__ __bf16 xs_t[2][64][40];

  const int stile = blockIdx.x;  // == h row of this spatial tile
  const int otile = blockIdx.y;
  const int b     = blockIdx.z;
  const int s0    = stile * 64;
  const int tid   = threadIdx.x;
  const int lane  = tid & 31;
  const int wave  = tid >> 5;
  const int n     = lane & 15;
  const int hi    = lane >> 4;

  const __bf16* arow = wt + (size_t)(otile * 64 + wave * 16 + n) * C_;

  v8f acc[4];
  zero_acc(acc);

  const int ldrow  = tid >> 2;  // k-local 0..31
  const int ldpart = tid & 3;   // s-quarter 0..3
  const float* xsrc =
      x + ((size_t)b * C_ + ldrow) * S_ + s0 + ldpart * 16;

  // prologue: stage chunk 0 into buffer 0
  {
    v4f v0 = *reinterpret_cast<const v4f*>(xsrc);
    v4f v1 = *reinterpret_cast<const v4f*>(xsrc + 4);
    v4f v2 = *reinterpret_cast<const v4f*>(xsrc + 8);
    v4f v3 = *reinterpret_cast<const v4f*>(xsrc + 12);
#pragma unroll
    for (int j = 0; j < 4; ++j) {
      xs_t[0][ldpart * 16 + j +  0][ldrow] = (__bf16)v0[j];
      xs_t[0][ldpart * 16 + j +  4][ldrow] = (__bf16)v1[j];
      xs_t[0][ldpart * 16 + j +  8][ldrow] = (__bf16)v2[j];
      xs_t[0][ldpart * 16 + j + 12][ldrow] = (__bf16)v3[j];
    }
  }
  __syncthreads();

  for (int it = 0; it < C_ / 32; ++it) {
    const int k0  = it * 32;
    const int cur = it & 1;
    const bool more = (it + 1) < C_ / 32;

    // issue next chunk's global loads first (latency hidden by WMMAs below)
    v4f n0, n1, n2, n3;
    if (more) {
      const float* src = xsrc + (size_t)(k0 + 32) * S_;
      __builtin_prefetch(src + 32 * S_, 0, 1);  // chunk k0+64 -> L2
      n0 = *reinterpret_cast<const v4f*>(src);
      n1 = *reinterpret_cast<const v4f*>(src + 4);
      n2 = *reinterpret_cast<const v4f*>(src + 8);
      n3 = *reinterpret_cast<const v4f*>(src + 12);
    }

    // batch all fragment loads, then a 4-deep WMMA burst
    v16bf a = ld16_bf(arow + k0 + hi * 16);
    v16bf bf[4];
#pragma unroll
    for (int j2 = 0; j2 < 4; ++j2) {
      union { v16bf v; v8bf h[2]; } u;
      const __bf16* bp = &xs_t[cur][j2 * 16 + n][hi * 16];
      u.h[0] = *reinterpret_cast<const v8bf*>(bp);
      u.h[1] = *reinterpret_cast<const v8bf*>(bp + 8);
      bf[j2] = u.v;
    }
#pragma unroll
    for (int j2 = 0; j2 < 4; ++j2) acc[j2] = wmma_bf16(a, bf[j2], acc[j2]);

    // stage next chunk into the other buffer
    if (more) {
      const int nb = cur ^ 1;
#pragma unroll
      for (int j = 0; j < 4; ++j) {
        xs_t[nb][ldpart * 16 + j +  0][ldrow] = (__bf16)n0[j];
        xs_t[nb][ldpart * 16 + j +  4][ldrow] = (__bf16)n1[j];
        xs_t[nb][ldpart * 16 + j +  8][ldrow] = (__bf16)n2[j];
        xs_t[nb][ldpart * 16 + j + 12][ldrow] = (__bf16)n3[j];
      }
    }
    __syncthreads();
  }

  // D layout: lane l, VGPR r -> M = r + 8*(l>>4), N = l&15
#pragma unroll
  for (int j2 = 0; j2 < 4; ++j2) {
    const int wcol = j2 * 16 + n;
    const int s    = s0 + wcol;
#pragma unroll
    for (int r = 0; r < 8; ++r) {
      const int m  = r + 8 * hi;
      const int ol = wave * 16 + m;
      float val = acc[j2][r];
      if (otile == 0) {
        val += bq[ol];
        qws[((size_t)b * S_ + s) * CQ_ + ol] = (__bf16)val;
      } else if (otile == 1) {
        val += bk[ol];
        kws[((size_t)b * S_ + s) * CQ_ + ol] = (__bf16)val;
      } else {
        const int c = (otile - 2) * 64 + ol;
        const __bf16 bv16 = (__bf16)(val + bv[c]);
        vws [((size_t)b * C_ + c) * S_ + s]                 = bv16;  // bchw
        vtws[((size_t)b * C_ + c) * S_ + wcol * H_ + stile] = bv16;  // bcwh
      }
    }
  }
}

// ---------------------------------------------------------------------------
// Kernel 2: attention logits (both branches).
// grid = (64, 2, B); block = 128 (4 waves).
// ---------------------------------------------------------------------------
__global__ __launch_bounds__(128) void logits_kernel(
    const __bf16* __restrict__ qws, const __bf16* __restrict__ kws,
    float* __restrict__ logits) {
  const int fixed = blockIdx.x;
  const int mode  = blockIdx.y;
  const int b     = blockIdx.z;
  const int lane  = threadIdx.x & 31;
  const int wave  = threadIdx.x >> 5;
  const int n     = lane & 15;
  const int hi    = lane >> 4;

  v8f acc[4];
  zero_acc(acc);

  const int arow = wave * 16 + n;

#pragma unroll
  for (int k0 = 0; k0 < CQ_; k0 += 32) {
    const int koff = k0 + hi * 16;
    v16bf a, bf[4];
    if (mode == 0) {
      a = ld16_bf(qws + ((size_t)b * S_ + arow * W_ + fixed) * CQ_ + koff);
#pragma unroll
      for (int j2 = 0; j2 < 4; ++j2)
        bf[j2] = ld16_bf(kws +
            ((size_t)b * S_ + (j2 * 16 + n) * W_ + fixed) * CQ_ + koff);
    } else {
      a = ld16_bf(qws + ((size_t)b * S_ + fixed * W_ + arow) * CQ_ + koff);
#pragma unroll
      for (int j2 = 0; j2 < 4; ++j2)
        bf[j2] = ld16_bf(kws +
            ((size_t)b * S_ + fixed * W_ + (j2 * 16 + n)) * CQ_ + koff);
    }
#pragma unroll
    for (int j2 = 0; j2 < 4; ++j2) acc[j2] = wmma_bf16(a, bf[j2], acc[j2]);
  }

#pragma unroll
  for (int j2 = 0; j2 < 4; ++j2) {
#pragma unroll
    for (int r = 0; r < 8; ++r) {
      const int m   = r + 8 * hi;
      const int row = wave * 16 + m;
      const int col = j2 * 16 + n;
      float val = acc[j2][r];
      if (mode == 0) {
        if (col == row) val = NEG_INF_;
        logits[((size_t)b * S_ + row * W_ + fixed) * 128 + col] = val;
      } else {
        logits[((size_t)b * S_ + fixed * W_ + row) * 128 + 64 + col] = val;
      }
    }
  }
}

// ---------------------------------------------------------------------------
// Kernel 3: softmax over 128 logits per (b,h,w); writes bf16 probabilities.
// ---------------------------------------------------------------------------
__global__ __launch_bounds__(256) void softmax_kernel(
    const float* __restrict__ logits, __bf16* __restrict__ probs) {
  const int lane = threadIdx.x & 31;
  const int wv   = threadIdx.x >> 5;
  const size_t row = (size_t)blockIdx.x * 8 + wv;
  const float* p = logits + row * 128;
  __bf16* q = probs + row * 128;

  float e[4];
#pragma unroll
  for (int j = 0; j < 4; ++j) e[j] = p[lane + 32 * j];

  float mx = fmaxf(fmaxf(e[0], e[1]), fmaxf(e[2], e[3]));
#pragma unroll
  for (int off = 16; off >= 1; off >>= 1)
    mx = fmaxf(mx, __shfl_xor(mx, off, 32));

  float sum = 0.f;
#pragma unroll
  for (int j = 0; j < 4; ++j) { e[j] = __expf(e[j] - mx); sum += e[j]; }
#pragma unroll
  for (int off = 16; off >= 1; off >>= 1) sum += __shfl_xor(sum, off, 32);

  const float inv = __builtin_amdgcn_rcpf(sum);
#pragma unroll
  for (int j = 0; j < 4; ++j) q[lane + 32 * j] = (__bf16)(e[j] * inv);
}

// ---------------------------------------------------------------------------
// Kernel 4: out_h = V * att_h^T per (b,w); writes out = x + gamma*out_h.
// grid = (C/64, W, B); block = 128 (4 waves).
// ---------------------------------------------------------------------------
__global__ __launch_bounds__(128) void out_h_kernel(
    const __bf16* __restrict__ vtws, const __bf16* __restrict__ probs,
    const float* __restrict__ x, const float* __restrict__ gamma,
    float* __restrict__ out) {
  const int ct   = blockIdx.x;
  const int w    = blockIdx.y;
  const int b    = blockIdx.z;
  const int lane = threadIdx.x & 31;
  const int wave = threadIdx.x >> 5;
  const int n    = lane & 15;
  const int hi   = lane >> 4;
  const int ca   = ct * 64 + wave * 16 + n;

  v8f acc[4];
  zero_acc(acc);

#pragma unroll
  for (int k0 = 0; k0 < H_; k0 += 32) {
    const int koff = k0 + hi * 16;  // g
    v16bf a =
        ld16_bf(vtws + ((size_t)b * C_ + ca) * S_ + (size_t)w * H_ + koff);
    v16bf bf[4];
#pragma unroll
    for (int j2 = 0; j2 < 4; ++j2)
      bf[j2] = ld16_bf(probs +
          ((size_t)b * S_ + (j2 * 16 + n) * W_ + w) * 128 + koff);
#pragma unroll
    for (int j2 = 0; j2 < 4; ++j2) acc[j2] = wmma_bf16(a, bf[j2], acc[j2]);
  }

  const float g = gamma[0];
#pragma unroll
  for (int j2 = 0; j2 < 4; ++j2) {
#pragma unroll
    for (int r = 0; r < 8; ++r) {
      const int m = r + 8 * hi;
      const int c = ct * 64 + wave * 16 + m;
      const int h = j2 * 16 + n;
      const size_t idx = ((size_t)b * C_ + c) * S_ + (size_t)h * W_ + w;
      out[idx] = x[idx] + g * acc[j2][r];
    }
  }
}

// ---------------------------------------------------------------------------
// Kernel 5: out_w = V * att_w^T per (b,h); accumulates out += gamma*out_w.
// grid = (C/64, H, B); block = 128 (4 waves).
// ---------------------------------------------------------------------------
__global__ __launch_bounds__(128) void out_w_kernel(
    const __bf16* __restrict__ vws, const __bf16* __restrict__ probs,
    const float* __restrict__ gamma, float* __restrict__ out) {
  const int ct   = blockIdx.x;
  const int h    = blockIdx.y;
  const int b    = blockIdx.z;
  const int lane = threadIdx.x & 31;
  const int wave = threadIdx.x >> 5;
  const int n    = lane & 15;
  const int hi   = lane >> 4;
  const int ca   = ct * 64 + wave * 16 + n;

  v8f acc[4];
  zero_acc(acc);

#pragma unroll
  for (int k0 = 0; k0 < W_; k0 += 32) {
    const int koff = k0 + hi * 16;  // u
    v16bf a =
        ld16_bf(vws + ((size_t)b * C_ + ca) * S_ + (size_t)h * W_ + koff);
    v16bf bf[4];
#pragma unroll
    for (int j2 = 0; j2 < 4; ++j2)
      bf[j2] = ld16_bf(probs +
          ((size_t)b * S_ + h * W_ + (j2 * 16 + n)) * 128 + 64 + koff);
#pragma unroll
    for (int j2 = 0; j2 < 4; ++j2) acc[j2] = wmma_bf16(a, bf[j2], acc[j2]);
  }

  const float g = gamma[0];
#pragma unroll
  for (int j2 = 0; j2 < 4; ++j2) {
#pragma unroll
    for (int r = 0; r < 8; ++r) {
      const int m  = r + 8 * hi;
      const int c  = ct * 64 + wave * 16 + m;
      const int wc = j2 * 16 + n;
      const size_t idx = ((size_t)b * C_ + c) * S_ + (size_t)h * W_ + wc;
      out[idx] += g * acc[j2][r];
    }
  }
}

// ---------------------------------------------------------------------------
// Launch: weights->bf16, qkv, logits, softmax, out_h (writes), out_w (accum).
// Workspace (~202 MB): q/k bf16 8.4 MB each, v bchw + vt bcwh bf16 67.1 MB
// each, wt bf16 0.7 MB, probs bf16 16.8 MB, logits f32 33.6 MB.
// ---------------------------------------------------------------------------
extern "C" void kernel_launch(void* const* d_in, const int* in_sizes, int n_in,
                              void* d_out, int out_size, void* d_ws,
                              size_t ws_size, hipStream_t stream) {
  const float* x     = (const float*)d_in[0];
  const float* Wq    = (const float*)d_in[1];
  const float* bq    = (const float*)d_in[2];
  const float* Wk    = (const float*)d_in[3];
  const float* bk    = (const float*)d_in[4];
  const float* Wv    = (const float*)d_in[5];
  const float* bv    = (const float*)d_in[6];
  const float* gamma = (const float*)d_in[7];
  float* out = (float*)d_out;

  __bf16* qws   = (__bf16*)d_ws;
  __bf16* kws   = qws + (size_t)B_ * S_ * CQ_;
  __bf16* vws   = kws + (size_t)B_ * S_ * CQ_;
  __bf16* vtws  = vws + (size_t)B_ * C_ * S_;
  __bf16* wt    = vtws + (size_t)B_ * C_ * S_;
  __bf16* probs = wt + (size_t)640 * C_;
  float*  logit = (float*)(probs + (size_t)B_ * S_ * 128);

  prep_weights_kernel<<<dim3((640 * C_) / (256 * 4)), 256, 0, stream>>>(
      Wq, Wk, Wv, wt);
  qkv_proj_kernel<<<dim3(S_ / 64, 10, B_), 128, 0, stream>>>(
      x, wt, bq, bk, bv, qws, kws, vws, vtws);
  logits_kernel<<<dim3(64, 2, B_), 128, 0, stream>>>(qws, kws, logit);
  softmax_kernel<<<dim3((B_ * S_) / 8), 256, 0, stream>>>(logit, probs);
  out_h_kernel<<<dim3(C_ / 64, W_, B_), 128, 0, stream>>>(vtws, probs, x,
                                                          gamma, out);
  out_w_kernel<<<dim3(C_ / 64, H_, B_), 128, 0, stream>>>(vws, probs, gamma,
                                                          out);
}
